// ChunkedCrossAttention_66434554134529
// MI455X (gfx1250) — matrix operand order repaired and checked by reading, not compiled, not run on hardware
//
#include <hip/hip_runtime.h>

// ---------------------------------------------------------------------------
// ChunkedCrossAttention for MI455X (gfx1250), bf16 WMMA pipeline.
// B=4 S=2048 C=32 N=2 L=128 HIDDEN=1024 NUM_HEADS=16 HEAD=64 CHUNK=64
//
// Projections: LDS-tiled double-buffered bf16 GEMM (64x128 block tile, k=32
// steps, padded LDS rows for conflict-free ds_load_b128; global loads for
// step k+1 issued before the WMMA burst of step k; all fragment ds_loads of
// a k-step hoisted ahead of the WMMA burst so LDS and XDL pipes overlap).
// Attention: per-(b,c,head) block, WMMA QK^T -> register softmax (shfl
// reductions) -> bf16 P via LDS relayout -> WMMA PV -> shifted store.
// ---------------------------------------------------------------------------

typedef __attribute__((ext_vector_type(16))) __bf16 v16bf;
typedef __attribute__((ext_vector_type(8)))  float  v8f;

#define HID   1024
#define NH    16
#define HD    64
#define CHK   64
#define CC    32
#define NB    4
#define NL    256   // N*L
#define SEQ   2048
#define LS    40    // padded LDS row stride (elements) for 32-wide k tiles

// ---- fragment layouts (contiguous k order used here; A and B match) ------------------------------
// A 16x32 bf16: lane half h, element e -> K = 16h + e
// B 32x16 bf16: lane half h, element e -> K = 16h + e
// C/D 16x16 f32: vgpr r, lane half h   -> M = r + 8h, N = lane&15

__device__ __forceinline__ v16bf afrag_bf(const __bf16* p, int k0, int half) {
    union { uint4 u[2]; v16bf v; } x;
    x.u[0] = *(const uint4*)(p + k0 + 16 * half);
    x.u[1] = *(const uint4*)(p + k0 + 16 * half + 8);
    return x.v;
}

__device__ __forceinline__ v16bf bfrag_bf(const __bf16* p, int k0, int half) {
    union { uint4 u[2]; v16bf v; } x;
    x.u[0] = *(const uint4*)(p + k0 + 16 * half);
    x.u[1] = *(const uint4*)(p + k0 + 16 * half + 8);
    return x.v;
}

// pack 8 fp32 -> 8 bf16 (one uint4)
__device__ __forceinline__ uint4 pack8(const float* f) {
    union { uint4 u; __bf16 h[8]; } x;
#pragma unroll
    for (int e = 0; e < 8; ++e) x.h[e] = (__bf16)f[e];
    return x.u;
}

#define WMMA_BF16(A, B, C) \
    __builtin_amdgcn_wmma_f32_16x16x32_bf16(false, (A), false, (B), (short)0, (C), false, false)

// ---------------------------------------------------------------------------
// Kernel 1: Q projection.  q_pad[b, c*64+i] = query[b, c*64+i+63] (0 past end)
// Qb[b][c][h][i][d] = bf16( q_pad @ Wq^T + bq )
// Block: 256 thr = 8 waves (4M x 2N), tile 64M x 128N, LDS double buffer.
// ---------------------------------------------------------------------------
__global__ __launch_bounds__(256) void qproj_kernel(
    const float* __restrict__ query, const float* __restrict__ Wq,
    const float* __restrict__ bq, __bf16* __restrict__ Qb)
{
    __shared__ __bf16 lds[2][64 * LS + 128 * LS];   // 30.7 KB

    const int tid  = threadIdx.x;
    const int w    = tid >> 5;
    const int lane = tid & 31;
    const int half = lane >> 4;
    const int l16  = lane & 15;

    const int m_base = blockIdx.x * 64;
    const int n_base = blockIdx.y * 128;

    // --- staging roles ---
    const int arow = tid >> 2, ak = (tid & 3) * 8;        // A tile 64x32
    const int am = m_base + arow;
    const int ab = am >> 11, at = am & 2047, as_ = at + 63;
    const float* aglob = (as_ < SEQ) ? (query + ((size_t)ab * SEQ + as_) * HID + ak) : nullptr;
    const int wrow = tid >> 1, wk = (tid & 1) * 16;       // W tile 128x32
    const float* wglob = Wq + (size_t)(n_base + wrow) * HID + wk;

    const int wm = w & 3, wn = w >> 2;

    v8f zero = {};
    v8f acc[4];
#pragma unroll
    for (int nt = 0; nt < 4; ++nt) acc[nt] = zero;

    alignas(16) float aR[8];
    alignas(16) float wR[16];

#pragma unroll
    for (int e = 0; e < 8; ++e) aR[e] = 0.0f;
    if (aglob) {
        *(float4*)(aR + 0) = *(const float4*)(aglob + 0);
        *(float4*)(aR + 4) = *(const float4*)(aglob + 4);
    }
#pragma unroll
    for (int q4 = 0; q4 < 4; ++q4)
        *(float4*)(wR + q4 * 4) = *(const float4*)(wglob + q4 * 4);

    int buf = 0;
    for (int ks = 0; ks < 32; ++ks) {
        __bf16* L = &lds[buf][0];
        *(uint4*)(L + arow * LS + ak)               = pack8(aR);
        *(uint4*)(L + 64 * LS + wrow * LS + wk)     = pack8(wR);
        *(uint4*)(L + 64 * LS + wrow * LS + wk + 8) = pack8(wR + 8);
        __syncthreads();

        // issue next k-step global loads before the WMMA burst
        if (ks < 31) {
            const int k1 = (ks + 1) * 32;
            if (aglob) {
                *(float4*)(aR + 0) = *(const float4*)(aglob + k1 + 0);
                *(float4*)(aR + 4) = *(const float4*)(aglob + k1 + 4);
            }
#pragma unroll
            for (int q4 = 0; q4 < 4; ++q4)
                *(float4*)(wR + q4 * 4) = *(const float4*)(wglob + k1 + q4 * 4);
        }

        // hoist ALL fragment loads ahead of the WMMA burst
        v16bf a = afrag_bf(L + (wm * 16 + l16) * LS, 0, half);
        v16bf bfr[4];
#pragma unroll
        for (int nt = 0; nt < 4; ++nt)
            bfr[nt] = bfrag_bf(L + 64 * LS + (wn * 64 + nt * 16 + l16) * LS, 0, half);
#pragma unroll
        for (int nt = 0; nt < 4; ++nt)
            acc[nt] = WMMA_BF16(a, bfr[nt], acc[nt]);

        buf ^= 1;
    }

    // epilogue: bias + store in [b][c][h][i][d] layout
    const int m0 = m_base + wm * 16;
    const int n0 = n_base + wn * 64;
#pragma unroll
    for (int nt = 0; nt < 4; ++nt) {
        const int n = n0 + nt * 16 + l16;
        const float bias = bq[n];
        const int hh = n >> 6, d = n & 63;
#pragma unroll
        for (int r = 0; r < 8; ++r) {
            const int mm = m0 + r + 8 * half;
            const int bb = mm >> 11;
            const int tt = mm & 2047;
            const int c  = tt >> 6, i = tt & 63;
            const size_t off = ((((size_t)bb * CC + c) * NH + hh) * CHK + i) * HD + d;
            Qb[off] = (__bf16)(acc[nt][r] + bias);
        }
    }
}

// ---------------------------------------------------------------------------
// Kernel 2: fused K/V projection (shared A tile, two W tiles in LDS).
// Kb[b][c][h][j][d] = bf16(kv @ Wk^T + bk),  Vt[b][c][h][d][j] = bf16(kv @ Wv^T + bv)
// ---------------------------------------------------------------------------
__global__ __launch_bounds__(256) void kvproj_kernel(
    const float* __restrict__ kv,
    const float* __restrict__ Wk, const float* __restrict__ bk,
    const float* __restrict__ Wv, const float* __restrict__ bv,
    __bf16* __restrict__ Kb, __bf16* __restrict__ Vt)
{
    __shared__ __bf16 lds[2][64 * LS + 2 * 128 * LS];   // 51.2 KB

    const int tid  = threadIdx.x;
    const int w    = tid >> 5;
    const int lane = tid & 31;
    const int half = lane >> 4;
    const int l16  = lane & 15;

    const int m_base = blockIdx.x * 64;
    const int n_base = blockIdx.y * 128;

    const int arow = tid >> 2, ak = (tid & 3) * 8;
    const float* aglob = kv + (size_t)(m_base + arow) * HID + ak;
    const int wrow = tid >> 1, wk = (tid & 1) * 16;
    const float* kglob = Wk + (size_t)(n_base + wrow) * HID + wk;
    const float* vglob = Wv + (size_t)(n_base + wrow) * HID + wk;

    const int wm = w & 3, wn = w >> 2;

    v8f zero = {};
    v8f accK[4], accV[4];
#pragma unroll
    for (int nt = 0; nt < 4; ++nt) { accK[nt] = zero; accV[nt] = zero; }

    alignas(16) float aR[8];
    alignas(16) float kR[16];
    alignas(16) float vR[16];

    *(float4*)(aR + 0) = *(const float4*)(aglob + 0);
    *(float4*)(aR + 4) = *(const float4*)(aglob + 4);
#pragma unroll
    for (int q4 = 0; q4 < 4; ++q4) {
        *(float4*)(kR + q4 * 4) = *(const float4*)(kglob + q4 * 4);
        *(float4*)(vR + q4 * 4) = *(const float4*)(vglob + q4 * 4);
    }

    const int WKOFF = 64 * LS;
    const int WVOFF = 64 * LS + 128 * LS;

    int buf = 0;
    for (int ks = 0; ks < 32; ++ks) {
        __bf16* L = &lds[buf][0];
        *(uint4*)(L + arow * LS + ak)             = pack8(aR);
        *(uint4*)(L + WKOFF + wrow * LS + wk)     = pack8(kR);
        *(uint4*)(L + WKOFF + wrow * LS + wk + 8) = pack8(kR + 8);
        *(uint4*)(L + WVOFF + wrow * LS + wk)     = pack8(vR);
        *(uint4*)(L + WVOFF + wrow * LS + wk + 8) = pack8(vR + 8);
        __syncthreads();

        if (ks < 31) {
            const int k1 = (ks + 1) * 32;
            *(float4*)(aR + 0) = *(const float4*)(aglob + k1 + 0);
            *(float4*)(aR + 4) = *(const float4*)(aglob + k1 + 4);
#pragma unroll
            for (int q4 = 0; q4 < 4; ++q4) {
                *(float4*)(kR + q4 * 4) = *(const float4*)(kglob + k1 + q4 * 4);
                *(float4*)(vR + q4 * 4) = *(const float4*)(vglob + k1 + q4 * 4);
            }
        }

        // hoist ALL fragment loads ahead of the WMMA burst
        v16bf a = afrag_bf(L + (wm * 16 + l16) * LS, 0, half);
        v16bf bkf[4], bvf[4];
#pragma unroll
        for (int nt = 0; nt < 4; ++nt) {
            const int brow = (wn * 64 + nt * 16 + l16) * LS;
            bkf[nt] = bfrag_bf(L + WKOFF + brow, 0, half);
            bvf[nt] = bfrag_bf(L + WVOFF + brow, 0, half);
        }
#pragma unroll
        for (int nt = 0; nt < 4; ++nt) {
            accK[nt] = WMMA_BF16(a, bkf[nt], accK[nt]);
            accV[nt] = WMMA_BF16(a, bvf[nt], accV[nt]);
        }

        buf ^= 1;
    }

    const int m0 = m_base + wm * 16;
    const int n0 = n_base + wn * 64;
#pragma unroll
    for (int nt = 0; nt < 4; ++nt) {
        const int n = n0 + nt * 16 + l16;
        const float biasK = bk[n], biasV = bv[n];
        const int hh = n >> 6, d = n & 63;
#pragma unroll
        for (int r = 0; r < 8; ++r) {
            const int mm  = m0 + r + 8 * half;
            const int bb  = mm >> 13;           // 8192 rows per batch
            const int rem = mm & 8191;
            const int c   = rem >> 8;           // 256 rows per chunk
            const int j   = rem & 255;
            const size_t head = (((size_t)bb * CC + c) * NH + hh);
            Kb[(head * NL + j) * HD + d] = (__bf16)(accK[nt][r] + biasK);
            Vt[(head * HD + d) * NL + j] = (__bf16)(accV[nt][r] + biasV);
        }
    }
}

// ---------------------------------------------------------------------------
// Kernel 3: attention per (b, c, head). Block = 128 thr = 4 waves, one 16-row
// q-tile per wave. S = Q K^T / 8 -> softmax (register + shfl) -> P in LDS
// (C-layout -> A-layout reshape) -> P @ V -> shifted store.
// ---------------------------------------------------------------------------
__global__ __launch_bounds__(128) void attn_kernel(
    const __bf16* __restrict__ Qb, const __bf16* __restrict__ Kb,
    const __bf16* __restrict__ Vt, float* __restrict__ out)
{
    __shared__ __bf16 Pl[4][16][NL];   // 32 KB

    const int w    = threadIdx.x >> 5;
    const int lane = threadIdx.x & 31;
    const int half = lane >> 4;
    const int l16  = lane & 15;

    const int bid = blockIdx.x;
    const int hh  = bid & 15;
    const int c   = (bid >> 4) & 31;
    const int b   = bid >> 9;

    const size_t head = (((size_t)b * CC + c) * NH + hh);
    const __bf16* Qh = Qb + head * (CHK * HD);
    const __bf16* Kh = Kb + head * (NL * HD);
    const __bf16* Vh = Vt + head * (HD * NL);

    const int i0 = w * 16;

    // ---- S = Q K^T  (16 x 256 per wave) ----
    v8f zero = {};
    v8f accS[16];
#pragma unroll
    for (int jt = 0; jt < 16; ++jt) accS[jt] = zero;

    const __bf16* qrow = Qh + (size_t)(i0 + l16) * HD;
    v16bf aQ0 = afrag_bf(qrow, 0, half);
    v16bf aQ1 = afrag_bf(qrow, 32, half);

    // process jt-tiles in pairs: 4 fragment loads, then 4 WMMAs
#pragma unroll
    for (int jp = 0; jp < 8; ++jp) {
        const __bf16* krow0 = Kh + (size_t)((2 * jp + 0) * 16 + l16) * HD;
        const __bf16* krow1 = Kh + (size_t)((2 * jp + 1) * 16 + l16) * HD;
        v16bf b00 = bfrag_bf(krow0, 0, half);
        v16bf b01 = bfrag_bf(krow0, 32, half);
        v16bf b10 = bfrag_bf(krow1, 0, half);
        v16bf b11 = bfrag_bf(krow1, 32, half);
        accS[2 * jp + 0] = WMMA_BF16(aQ0, b00, accS[2 * jp + 0]);
        accS[2 * jp + 0] = WMMA_BF16(aQ1, b01, accS[2 * jp + 0]);
        accS[2 * jp + 1] = WMMA_BF16(aQ0, b10, accS[2 * jp + 1]);
        accS[2 * jp + 1] = WMMA_BF16(aQ1, b11, accS[2 * jp + 1]);
    }

    // ---- softmax over j (scale 1/sqrt(64) = 0.125) ----
    const float sc = 0.125f;
#pragma unroll
    for (int r = 0; r < 8; ++r) {
        float mx = -3.0e30f;
#pragma unroll
        for (int jt = 0; jt < 16; ++jt) mx = fmaxf(mx, accS[jt][r]);
#pragma unroll
        for (int msk = 1; msk < 16; msk <<= 1) mx = fmaxf(mx, __shfl_xor(mx, msk, 32));
        float sum = 0.0f;
#pragma unroll
        for (int jt = 0; jt < 16; ++jt) {
            float e = __expf(sc * (accS[jt][r] - mx));
            accS[jt][r] = e;
            sum += e;
        }
#pragma unroll
        for (int msk = 1; msk < 16; msk <<= 1) sum += __shfl_xor(sum, msk, 32);
        const float inv = 1.0f / sum;
#pragma unroll
        for (int jt = 0; jt < 16; ++jt) accS[jt][r] *= inv;
    }

    // ---- spill P (bf16) to LDS: C-layout -> A-layout reshape ----
#pragma unroll
    for (int jt = 0; jt < 16; ++jt) {
#pragma unroll
        for (int r = 0; r < 8; ++r) {
            Pl[w][r + 8 * half][jt * 16 + l16] = (__bf16)accS[jt][r];
        }
    }
    __syncthreads();

    // ---- O = P @ V  (16 x 64 per wave) ----
    v8f accO[4];
#pragma unroll
    for (int dt = 0; dt < 4; ++dt) accO[dt] = zero;

    const __bf16* prow = &Pl[w][l16][0];
#pragma unroll
    for (int kt = 0; kt < 8; ++kt) {
        v16bf aP = afrag_bf(prow, kt * 32, half);
        v16bf bV[4];
#pragma unroll
        for (int dt = 0; dt < 4; ++dt)
            bV[dt] = bfrag_bf(Vh + (size_t)(dt * 16 + l16) * NL, kt * 32, half);
#pragma unroll
        for (int dt = 0; dt < 4; ++dt)
            accO[dt] = WMMA_BF16(aP, bV[dt], accO[dt]);
    }

    // ---- store with +63 shift: out[b, 63 + c*64 + i, h*64 + d] ----
#pragma unroll
    for (int dt = 0; dt < 4; ++dt) {
        const int col = hh * HD + dt * 16 + l16;
#pragma unroll
        for (int r = 0; r < 8; ++r) {
            const int i = i0 + r + 8 * half;
            const int s = 63 + c * CHK + i;
            if (s < SEQ) out[((size_t)b * SEQ + s) * HID + col] = accO[dt][r];
        }
    }
}

// ---------------------------------------------------------------------------
// Kernel 4: zero the 63 front-padded rows per batch.
// ---------------------------------------------------------------------------
__global__ __launch_bounds__(256) void zero_head_kernel(float* __restrict__ out)
{
    const int idx = blockIdx.x * 256 + threadIdx.x;
    const int per = 63 * HID;
    if (idx < NB * per) {
        const int b = idx / per;
        const int r = idx % per;
        out[(size_t)b * SEQ * HID + r] = 0.0f;
    }
}

// ---------------------------------------------------------------------------
extern "C" void kernel_launch(void* const* d_in, const int* in_sizes, int n_in,
                              void* d_out, int out_size, void* d_ws, size_t ws_size,
                              hipStream_t stream)
{
    const float* query = (const float*)d_in[0];
    const float* kv    = (const float*)d_in[1];
    const float* Wq    = (const float*)d_in[2];
    const float* bq    = (const float*)d_in[3];
    const float* Wk    = (const float*)d_in[4];
    const float* bk    = (const float*)d_in[5];
    const float* Wv    = (const float*)d_in[6];
    const float* bv    = (const float*)d_in[7];
    float* out = (float*)d_out;

    char* ws = (char*)d_ws;
    __bf16* Qb = (__bf16*)ws;                                        // 8192*1024*2  = 16 MB
    __bf16* Kb = (__bf16*)(ws + (size_t)16 * 1024 * 1024);           // 32768*1024*2 = 64 MB
    __bf16* Vt = (__bf16*)(ws + (size_t)80 * 1024 * 1024);           // 64 MB

    qproj_kernel<<<dim3(128, 8), 256, 0, stream>>>(query, Wq, bq, Qb);
    kvproj_kernel<<<dim3(512, 8), 256, 0, stream>>>(kv, Wk, bk, Wv, bv, Kb, Vt);
    attn_kernel<<<dim3(NB * CC * NH), 128, 0, stream>>>(Qb, Kb, Vt, out);
    zero_head_kernel<<<dim3((NB * 63 * HID + 255) / 256), 256, 0, stream>>>(out);
}
